// TopKPromptSelector_87643102642860
// MI455X (gfx1250) — matrix operand order repaired and checked by reading, not compiled, not run on hardware
//
#include <hip/hip_runtime.h>
#include <hip/hip_bf16.h>

// ---------------------------------------------------------------------------
// TopKPromptSelector fused kernel for gfx1250 (MI455X, wave32, WMMA).
//   scores = A[16384,768] @ W^T[768,8192] + b  -> per-row top-5 -> softmax
//   out    = sum_k softmax_k * prompt_pool[top_idx_k]
// bf16 WMMA GEMM (v_wmma_f32_16x16x32_bf16) fused with running top-5.
// A and W pre-converted to bf16 in d_ws (25.2 + 12.6 MB, L2-resident W).
// Both tiles double-buffered via GLOBAL_LOAD_ASYNC_TO_LDS_B128 (ASYNCcnt),
// one barrier per K-chunk.
// ---------------------------------------------------------------------------

typedef __bf16 v16bf __attribute__((ext_vector_type(16)));
typedef float  v8f   __attribute__((ext_vector_type(8)));
typedef unsigned int u32;
typedef unsigned long long u64;

struct alignas(16) U4 { u32 x, y, z, w; };

union ABFrag {
    v16bf v;
    U4    q[2];
};

#define NB       16384
#define DIMK     768
#define NPROMPT  8192
#define KCHUNKS  (DIMK / 64)   // 12

__device__ __forceinline__ u32 f2bf_bits(float f) {
    u32 u = __float_as_uint(f);
    u32 r = u + 0x7FFFu + ((u >> 16) & 1u);   // round-to-nearest-even
    return r >> 16;
}
__device__ __forceinline__ u32 pk2(float lo, float hi) {
    return f2bf_bits(lo) | (f2bf_bits(hi) << 16);
}

// sorted top-5 insert (descending). Strict '>' keeps earliest index on ties,
// matching jax.lax.top_k tie-breaking for an ascending scan.
__device__ __forceinline__ void insert5(float c, int ci, float s[5], int idx[5]) {
    if (c <= s[4]) return;
    if (c > s[0]) {
        s[4]=s[3]; idx[4]=idx[3]; s[3]=s[2]; idx[3]=idx[2];
        s[2]=s[1]; idx[2]=idx[1]; s[1]=s[0]; idx[1]=idx[0];
        s[0]=c; idx[0]=ci;
    } else if (c > s[1]) {
        s[4]=s[3]; idx[4]=idx[3]; s[3]=s[2]; idx[3]=idx[2];
        s[2]=s[1]; idx[2]=idx[1]; s[1]=c; idx[1]=ci;
    } else if (c > s[2]) {
        s[4]=s[3]; idx[4]=idx[3]; s[3]=s[2]; idx[3]=idx[2];
        s[2]=c; idx[2]=ci;
    } else if (c > s[3]) {
        s[4]=s[3]; idx[4]=idx[3]; s[3]=c; idx[3]=ci;
    } else {
        s[4]=c; idx[4]=ci;
    }
}

// ---------------------------------------------------------------------------
// Kernel 0: fp32 -> bf16 row-major copy into workspace (used for W and A).
// ---------------------------------------------------------------------------
__global__ __launch_bounds__(256)
void cvt_bf16(const float* __restrict__ src, unsigned short* __restrict__ dst, int n4) {
    int i = blockIdx.x * 256 + threadIdx.x;
    if (i >= n4) return;
    float4 f = ((const float4*)src)[i];
    u64 p = (u64)pk2(f.x, f.y) | ((u64)pk2(f.z, f.w) << 32);
    ((u64*)dst)[i] = p;
}

// ---------------------------------------------------------------------------
// Fused GEMM + bias + top-5 + softmax + gather.
// Block: 256 threads = 8 waves. 64 rows per block. Waves arranged 4x2:
//   wave w -> rows (w>>1)*16 .. +15, cols (w&1)*64 .. +63 of the 64x128 tile.
// ---------------------------------------------------------------------------
__global__ __launch_bounds__(256)
void fused_topk_prompt(const unsigned short* __restrict__ Ab,
                       const unsigned short* __restrict__ Wb,
                       const float* __restrict__ bias,
                       const float* __restrict__ pool,
                       float* __restrict__ out) {
    __shared__ unsigned short A_s[2][64][72];   // 18432 B (double-buffered)
    __shared__ unsigned short B_s[2][128][72];  // 36864 B (double-buffered)
    __shared__ float bias_s[128];               //   512 B
    __shared__ char  mrg[8704];                 //  8704 B  => 64512 B total
    // mrg overlay: during ng loop = per-wave stage [8][272] f32;
    // after loop  = smTop[64][2][5] f32 | smIdx[64][2][5] i32 | smW[64][5] | smI[64][5]
    float* stage          = (float*)mrg;
    float (*smTop)[2][5]  = (float(*)[2][5])(mrg);
    int   (*smIdx)[2][5]  = (int  (*)[2][5])(mrg + 2560);
    float (*smW)[5]       = (float(*)[5])(mrg + 5120);
    int   (*smI)[5]       = (int  (*)[5])(mrg + 6400);

    const int tid  = threadIdx.x;
    const int wid  = tid >> 5;
    const int lane = tid & 31;
    const int m    = lane & 15;
    const int half = lane >> 4;
    const int rowOff = (wid >> 1) << 4;        // 0,16,32,48
    const int colOff = (wid & 1) << 6;         // 0,64
    const int rowBase = blockIdx.x * 64;

    // cooperative async-DMA roles
    const int ar  = tid >> 2;                  // A row (0..63), 32 B per thread
    const int akq = (tid & 3) << 4;            // A k offset (0,16,32,48)
    const int bn  = tid & 127;                 // B prompt row (0..127), 64 B per thread
    const int bh  = tid >> 7;                  // B k-half (0/1)
    const u64 sbaseA = (u64)(size_t)(Ab + (size_t)rowBase * DIMK);

    float ts[5] = { -__builtin_inff(), -__builtin_inff(), -__builtin_inff(),
                    -__builtin_inff(), -__builtin_inff() };
    int   ti[5] = { 0, 0, 0, 0, 0 };

    const v8f vzero = {};

    // async DMA segments (GVS: mem = SADDR + VADDR32 + inst_off; LDS = VDST + inst_off)
    auto issueA = [&](int kc, int buf) {
        u32 ldsa = (u32)(size_t)&A_s[buf][ar][akq];
        u32 voff = (u32)(((u32)ar * DIMK + (u32)kc + (u32)akq) * 2u);
        asm volatile(
            "global_load_async_to_lds_b128 %0, %1, %2\n\t"
            "global_load_async_to_lds_b128 %0, %1, %2 offset:16"
            :: "v"(ldsa), "v"(voff), "s"(sbaseA) : "memory");
    };
    auto issueB = [&](u64 sbase, int kc, int buf) {
        u32 ldsa = (u32)(size_t)&B_s[buf][bn][bh * 32];
        u32 voff = (u32)(((u32)bn * DIMK + (u32)kc + (u32)bh * 32) * 2u);
        asm volatile(
            "global_load_async_to_lds_b128 %0, %1, %2\n\t"
            "global_load_async_to_lds_b128 %0, %1, %2 offset:16\n\t"
            "global_load_async_to_lds_b128 %0, %1, %2 offset:32\n\t"
            "global_load_async_to_lds_b128 %0, %1, %2 offset:48"
            :: "v"(ldsa), "v"(voff), "s"(sbase) : "memory");
    };

    for (int ng = 0; ng < NPROMPT; ng += 128) {
        __syncthreads();   // protect bias_s / tile buffers from prior-iter top-k readers

        v8f acc[4];
#pragma unroll
        for (int t = 0; t < 4; ++t) acc[t] = vzero;

        const u64 sbase = (u64)(size_t)(Wb + (size_t)ng * DIMK);

        // ---- prologue: chunk 0 in flight ----
        issueA(0, 0);
        issueB(sbase, 0, 0);
        if (tid < 128) bias_s[tid] = bias[ng + tid];
        asm volatile("s_wait_asynccnt 0" ::: "memory");
        __syncthreads();

        for (int ic = 0; ic < KCHUNKS; ++ic) {
            const int cur = ic & 1;
            if (ic + 1 < KCHUNKS) {            // next chunk: both tiles async
                issueA((ic + 1) * 64, cur ^ 1);
                issueB(sbase, (ic + 1) * 64, cur ^ 1);
            }

            // ---- compute chunk ic: 2 k-steps x 4 N-tiles = 8 WMMAs/wave ----
#pragma unroll
            for (int ks = 0; ks < 2; ++ks) {
                const int k0 = ks * 32;
                ABFrag a;
                // ISA 7.12.2 16-bit A 16x32: lane half0 K{0..7,16..23}, half1 +8
                a.q[0] = *(const U4*)&A_s[cur][rowOff + m][k0 + half * 8];
                a.q[1] = *(const U4*)&A_s[cur][rowOff + m][k0 + half * 8 + 16];
#pragma unroll
                for (int t = 0; t < 4; ++t) {
                    ABFrag b;
                    const unsigned short* bp = &B_s[cur][colOff + t * 16 + m][k0 + half * 8];
                    b.q[0] = ((const U4*)bp)[0];
                    b.q[1] = ((const U4*)bp)[2];
                    acc[t] = __builtin_amdgcn_wmma_f32_16x16x32_bf16(
                        false, a.v, false, b.v, (short)0, acc[t], false, false);
                }
            }

            asm volatile("s_wait_asynccnt 0" ::: "memory");  // next tiles landed
            __syncthreads();           // all waves done with buf[cur], next visible
        }

        // ---- stage each 16x16 tile through per-wave LDS, update top-5 ----
        float* st = stage + wid * 272;
#pragma unroll
        for (int t = 0; t < 4; ++t) {
#pragma unroll
            for (int v = 0; v < 8; ++v)
                st[(v + (half << 3)) * 17 + m] = acc[t][v];
            asm volatile("s_wait_dscnt 0" ::: "memory");
            if (lane < 16) {
                const float* rp = st + lane * 17;
                const int nb = colOff + t * 16;
#pragma unroll
                for (int j = 0; j < 16; ++j) {
                    float c = rp[j] + bias_s[nb + j];
                    insert5(c, ng + nb + j, ts, ti);
                }
            }
            asm volatile("" ::: "memory");  // keep next tile's stores below reads
        }
    }

    __syncthreads();   // staging region retires before merge overlay reuses it

    // ---- merge the two column-half top-5 lists per row ----
    if (lane < 16) {
        const int row = rowOff + lane;
        const int h   = wid & 1;
#pragma unroll
        for (int j = 0; j < 5; ++j) { smTop[row][h][j] = ts[j]; smIdx[row][h][j] = ti[j]; }
    }
    __syncthreads();

    if (tid < 64) {
        float os[5]; int oi[5];
#pragma unroll
        for (int j = 0; j < 5; ++j) { os[j] = smTop[tid][0][j]; oi[j] = smIdx[tid][0][j]; }
#pragma unroll
        for (int j = 0; j < 5; ++j) insert5(smTop[tid][1][j], smIdx[tid][1][j], os, oi);
        // softmax over the sorted 5 (os[0] is the max)
        float e[5], sum = 0.f;
#pragma unroll
        for (int j = 0; j < 5; ++j) { e[j] = __expf(os[j] - os[0]); sum += e[j]; }
        float inv = 1.f / sum;
#pragma unroll
        for (int j = 0; j < 5; ++j) { smW[tid][j] = e[j] * inv; smI[tid][j] = oi[j]; }
    }
    __syncthreads();

    // ---- gather: out[row] = sum_j w_j * pool[idx_j]  (float4, coalesced) ----
    for (int it = tid; it < 64 * (DIMK / 4); it += 256) {
        const int r  = it / (DIMK / 4);
        const int c4 = it % (DIMK / 4);
        const float w0 = smW[r][0], w1 = smW[r][1], w2 = smW[r][2],
                    w3 = smW[r][3], w4 = smW[r][4];
        const float4 p0 = ((const float4*)(pool + (size_t)smI[r][0] * DIMK))[c4];
        const float4 p1 = ((const float4*)(pool + (size_t)smI[r][1] * DIMK))[c4];
        const float4 p2 = ((const float4*)(pool + (size_t)smI[r][2] * DIMK))[c4];
        const float4 p3 = ((const float4*)(pool + (size_t)smI[r][3] * DIMK))[c4];
        const float4 p4 = ((const float4*)(pool + (size_t)smI[r][4] * DIMK))[c4];
        float4 o;
        o.x = w0*p0.x + w1*p1.x + w2*p2.x + w3*p3.x + w4*p4.x;
        o.y = w0*p0.y + w1*p1.y + w2*p2.y + w3*p3.y + w4*p4.y;
        o.z = w0*p0.z + w1*p1.z + w2*p2.z + w3*p3.z + w4*p4.z;
        o.w = w0*p0.w + w1*p1.w + w2*p2.w + w3*p3.w + w4*p4.w;
        ((float4*)(out + (size_t)(rowBase + r) * DIMK))[c4] = o;
    }
}

extern "C" void kernel_launch(void* const* d_in, const int* in_sizes, int n_in,
                              void* d_out, int out_size, void* d_ws, size_t ws_size,
                              hipStream_t stream) {
    (void)in_sizes; (void)n_in; (void)out_size; (void)ws_size;
    const float* A    = (const float*)d_in[0];   // [16384, 768]
    const float* W    = (const float*)d_in[1];   // [8192, 768]
    const float* bias = (const float*)d_in[2];   // [8192]
    const float* pool = (const float*)d_in[3];   // [8192, 768]
    float* out = (float*)d_out;                  // [16384, 768]

    unsigned short* Wb = (unsigned short*)d_ws;              // 12.6 MB
    unsigned short* Ab = Wb + (size_t)NPROMPT * DIMK;        // 25.2 MB

    const int n4W = (NPROMPT * DIMK) / 4;        // 1,572,864
    const int n4A = (NB * DIMK) / 4;             // 3,145,728
    cvt_bf16<<<(n4W + 255) / 256, 256, 0, stream>>>(W, Wb, n4W);
    cvt_bf16<<<(n4A + 255) / 256, 256, 0, stream>>>(A, Ab, n4A);
    fused_topk_prompt<<<NB / 64, 256, 0, stream>>>(Ab, Wb, bias, pool, out);
}
